// MHSA_14903536517198
// MI455X (gfx1250) — matrix operand order repaired and checked by reading, not compiled, not run on hardware
//
#include <hip/hip_runtime.h>

// ---------------------------------------------------------------------------
// MHSA (per-position attention over dk, contracting h) for MI455X / gfx1250
//   LN -> bf16 | W -> bf16 transposed | QKV GEMM: v_wmma_f32_16x16x32_bf16
//   with Tensor-Data-Mover (tensor_load_to_lds) double-buffered LDS staging
//   per-position softmax attention + residual in f32
// ---------------------------------------------------------------------------

#define D_MODEL 1024
#define DK      64
#define NH      16            // h dimension of the reshape
#define ROWS    8192          // B*M
#define LN_EPS  1e-5f

typedef __attribute__((ext_vector_type(16))) __bf16 v16bf;
typedef __attribute__((ext_vector_type(8)))  __bf16 v8bf;
typedef __attribute__((ext_vector_type(8)))  float  v8f;
typedef __attribute__((ext_vector_type(4)))  unsigned int v4u;
typedef __attribute__((ext_vector_type(8)))  int    v8i;
typedef __attribute__((ext_vector_type(4)))  int    v4i;

// round-to-nearest-even f32 -> bf16
__device__ __forceinline__ unsigned short f32_to_bf16(float f) {
    union { float f; unsigned int u; } v;
    v.f = f;
    unsigned int u = v.u;
    u += 0x7FFFu + ((u >> 16) & 1u);
    return (unsigned short)(u >> 16);
}

// low 32 bits of the generic address of an LDS location == raw LDS byte offset
// (flat->LDS mapping truncates to addr[31:0], ISA 10.2)
__device__ __forceinline__ unsigned lds_byte_off(const void* p) {
    return (unsigned)(unsigned long long)(uintptr_t)p;
}

// ---------------------------------------------------------------------------
// TDM: issue a 2D tile load global->LDS via tensor_load_to_lds.
// Builds D# groups per CDNA5 ISA 8.3/8.4 (bf16 elements, data_size code 1).
//   tile_x  : tile width in elements (contiguous, dim0)
//   tile_y  : tile height in rows    (dim1)
//   stride  : row stride of the global tensor in elements
// ---------------------------------------------------------------------------
__device__ __forceinline__ void tdm_load_tile_bf16(unsigned lds_off,
                                                   const void* gptr,
                                                   unsigned tile_x, unsigned tile_y,
                                                   unsigned tensor_x, unsigned tensor_y,
                                                   unsigned stride) {
    const unsigned long long ga = (unsigned long long)(uintptr_t)gptr;

    v4u g0;
    g0[0] = 1u;                                            // count=1, user descriptor
    g0[1] = lds_off;                                       // lds_addr (bytes)
    g0[2] = (unsigned)(ga & 0xFFFFFFFFu);                  // global_addr[31:0]
    g0[3] = (unsigned)((ga >> 32) & 0x01FFFFFFu)           // global_addr[56:32]
          | (2u << 30);                                    // type = 2 ("image")

    // group1 (256 bits) as 4x u64, then split
    const unsigned long long q0 =
        (1ull << 16)                                       // data_size = 1 (2 bytes)
      | ((unsigned long long)(tensor_x & 0xFFFFu) << 48);  // tensor_dim0[15:0]
    const unsigned long long q1 =
        ((unsigned long long)(tensor_x >> 16))             // tensor_dim0[31:16]
      | ((unsigned long long)tensor_y << 16)               // tensor_dim1
      | ((unsigned long long)(tile_x & 0xFFFFu) << 48);    // tile_dim0
    const unsigned long long q2 =
        ((unsigned long long)(tile_y & 0xFFFFu))           // tile_dim1
      | ((unsigned long long)stride << 32);                // tensor_dim0_stride[31:0]
    const unsigned long long q3 = 0ull;                    // stride hi / dim1_stride

    v8i g1;
    g1[0] = (int)(q0);       g1[1] = (int)(q0 >> 32);
    g1[2] = (int)(q1);       g1[3] = (int)(q1 >> 32);
    g1[4] = (int)(q2);       g1[5] = (int)(q2 >> 32);
    g1[6] = (int)(q3);       g1[7] = (int)(q3 >> 32);

    const v4i z4 = {0, 0, 0, 0};
#if defined(__clang_major__) && (__clang_major__ >= 23)
    const v8i z8 = {0, 0, 0, 0, 0, 0, 0, 0};
    __builtin_amdgcn_tensor_load_to_lds(g0, g1, z4, z4, z8, 0);
#else
    __builtin_amdgcn_tensor_load_to_lds(g0, g1, z4, z4, 0);
#endif
}

// ---------------------------------------------------------------------------
// Kernel 1: LayerNorm over last dim, output bf16 XN[row][d]
// ---------------------------------------------------------------------------
__global__ __launch_bounds__(256)
void ln_bf16_kernel(const float* __restrict__ inp,
                    const float* __restrict__ gamma,
                    const float* __restrict__ beta,
                    unsigned short* __restrict__ xn) {
    const int row = blockIdx.x;
    const int tid = threadIdx.x;
    const float* x = inp + (size_t)row * D_MODEL;

    float v[4];
    float s = 0.f;
#pragma unroll
    for (int i = 0; i < 4; ++i) { v[i] = x[tid + i * 256]; s += v[i]; }

    __shared__ float red[256];
    red[tid] = s;
    __syncthreads();
    for (int off = 128; off > 0; off >>= 1) {
        if (tid < off) red[tid] += red[tid + off];
        __syncthreads();
    }
    const float mu = red[0] * (1.0f / D_MODEL);
    __syncthreads();

    float s2 = 0.f;
#pragma unroll
    for (int i = 0; i < 4; ++i) { float d = v[i] - mu; s2 += d * d; }
    red[tid] = s2;
    __syncthreads();
    for (int off = 128; off > 0; off >>= 1) {
        if (tid < off) red[tid] += red[tid + off];
        __syncthreads();
    }
    const float var  = red[0] * (1.0f / D_MODEL);
    const float rstd = rsqrtf(var + LN_EPS);

    unsigned short* dst = xn + (size_t)row * D_MODEL;
#pragma unroll
    for (int i = 0; i < 4; ++i) {
        const int d = tid + i * 256;
        const float y = (v[i] - mu) * rstd * gamma[d] + beta[d];
        dst[d] = f32_to_bf16(y);
    }
}

// ---------------------------------------------------------------------------
// Kernel 2: convert W[k][n] (f32) -> Wt[w][n][k] (bf16, transposed)
// ---------------------------------------------------------------------------
__global__ __launch_bounds__(256)
void wconv_kernel(const float* __restrict__ Wq,
                  const float* __restrict__ Wk,
                  const float* __restrict__ Wv,
                  unsigned short* __restrict__ wt) {
    const int w = blockIdx.z;
    const float* W = (w == 0) ? Wq : ((w == 1) ? Wk : Wv);
    unsigned short* dst = wt + (size_t)w * D_MODEL * D_MODEL;

    __shared__ float tile[32][33];
    const int n0 = blockIdx.x * 32;
    const int k0 = blockIdx.y * 32;
    const int tx = threadIdx.x;       // 0..31
    const int ty = threadIdx.y;       // 0..7

#pragma unroll
    for (int i = 0; i < 32; i += 8)
        tile[ty + i][tx] = W[(size_t)(k0 + ty + i) * D_MODEL + (n0 + tx)];
    __syncthreads();
#pragma unroll
    for (int i = 0; i < 32; i += 8)
        dst[(size_t)(n0 + ty + i) * D_MODEL + (k0 + tx)] =
            f32_to_bf16(tile[tx][ty + i]);
}

// ---------------------------------------------------------------------------
// Kernel 3: QKV GEMM   out[w][row][n] = XN[row][:] . Wt[w][n][:] + bias[w][n]
// 256 threads = 8 wave32s; macro tile 128(M) x 128(N); K-step 32.
// Each wave owns a 32x64 output tile = 2x4 v_wmma_f32_16x16x32_bf16.
// A/B tiles staged by the Tensor Data Mover (double-buffered LDS),
// synchronized with s_wait_tensorcnt + workgroup barrier.
// ---------------------------------------------------------------------------
__global__ __launch_bounds__(256)
void qkv_gemm_kernel(const unsigned short* __restrict__ xn,
                     const unsigned short* __restrict__ wt,
                     const float* __restrict__ bq,
                     const float* __restrict__ bk,
                     const float* __restrict__ bv,
                     float* __restrict__ qkv) {
    const int w = blockIdx.z;
    const float* bias = (w == 0) ? bq : ((w == 1) ? bk : bv);
    const unsigned short* W = wt + (size_t)w * D_MODEL * D_MODEL;
    float* out = qkv + (size_t)w * ROWS * D_MODEL;

    // double-buffered tiles: A 128x32 bf16, B(t) 128x32 bf16 -> 4 x 8KB = 32KB
    __shared__ unsigned short Alds[2][128 * 32];
    __shared__ unsigned short Blds[2][128 * 32];

    const int tid  = threadIdx.x;
    const int lane = tid & 31;
    const int wave = tid >> 5;        // 0..7
    const int wm   = wave >> 1;       // 4 row groups of 32
    const int wn   = wave & 1;        // 2 col groups of 64
    const int row0 = blockIdx.y * 128;
    const int col0 = blockIdx.x * 128;

    const int r16 = lane & 15;
    const int kh  = lane >> 4;        // which K half-group this lane holds

    // prologue: TDM-load K-tile 0 into buffer 0 (one issuing thread/wave)
    if (tid == 0) {
        tdm_load_tile_bf16(lds_byte_off(&Alds[0][0]),
                           &xn[(size_t)row0 * D_MODEL],
                           32, 128, D_MODEL, ROWS, D_MODEL);
        tdm_load_tile_bf16(lds_byte_off(&Blds[0][0]),
                           &W[(size_t)col0 * D_MODEL],
                           32, 128, D_MODEL, D_MODEL, D_MODEL);
    }
    __builtin_amdgcn_s_wait_tensorcnt(0);   // non-issuing waves pass instantly
    __syncthreads();

    v8f acc[2][4] = {};

    for (int k0 = 0; k0 < D_MODEL; k0 += 32) {
        const int cur = (k0 >> 5) & 1;
        const int nxt = cur ^ 1;

        // issue next K-tile DMA while computing on the current one
        if ((k0 + 32 < D_MODEL) && (tid == 0)) {
            tdm_load_tile_bf16(lds_byte_off(&Alds[nxt][0]),
                               &xn[(size_t)row0 * D_MODEL + k0 + 32],
                               32, 128, D_MODEL, ROWS, D_MODEL);
            tdm_load_tile_bf16(lds_byte_off(&Blds[nxt][0]),
                               &W[(size_t)col0 * D_MODEL + k0 + 32],
                               32, 128, D_MODEL, D_MODEL, D_MODEL);
        }
        // L2 prefetch two tiles ahead
        __builtin_prefetch(&xn[(size_t)(row0 + (tid >> 1)) * D_MODEL +
                               ((k0 + 64 < D_MODEL) ? (k0 + 64) : k0)], 0, 1);

        // ---- fragments in ISA bf16 layouts ----
        // A 16x32: lane(<16) row=r16 holds K[kh*8..+8) and K[16+kh*8..+8)
        // B 32x16: lane(<16) col=r16 holds K[kh*8..+8) and K[16+kh*8..+8)
        union Frag { v16bf v; v8bf h[2]; } a[2], b[4];
#pragma unroll
        for (int t = 0; t < 2; ++t) {
            const int ar = wm * 32 + t * 16 + r16;
            a[t].h[0] = *(const v8bf*)&Alds[cur][ar * 32 + kh * 8];
            a[t].h[1] = *(const v8bf*)&Alds[cur][ar * 32 + 16 + kh * 8];
        }
#pragma unroll
        for (int u = 0; u < 4; ++u) {
            const int bc = wn * 64 + u * 16 + r16;
            b[u].h[0] = *(const v8bf*)&Blds[cur][bc * 32 + kh * 8];
            b[u].h[1] = *(const v8bf*)&Blds[cur][bc * 32 + 16 + kh * 8];
        }

#pragma unroll
        for (int t = 0; t < 2; ++t)
#pragma unroll
            for (int u = 0; u < 4; ++u)
                acc[t][u] = __builtin_amdgcn_wmma_f32_16x16x32_bf16(
                    /*neg_a=*/false, a[t].v,
                    /*neg_b=*/false, b[u].v,
                    /*c_mod=*/(short)0, acc[t][u],
                    /*reuse_a=*/false, /*reuse_b=*/false);

        __builtin_amdgcn_s_wait_tensorcnt(0);  // next buffer landed (wave 0)
        __syncthreads();                       // publish to all waves
    }

    // ---- epilogue: C/D layout -> row = t*16 + (lane>=16)*8 + e, col = lane&15
    const int colin  = lane & 15;
    const int rowsel = lane >> 4;
#pragma unroll
    for (int t = 0; t < 2; ++t) {
#pragma unroll
        for (int u = 0; u < 4; ++u) {
            const int ccol = col0 + wn * 64 + u * 16 + colin;
            const float bb = bias[ccol];
#pragma unroll
            for (int e = 0; e < 8; ++e) {
                const int crow = row0 + wm * 32 + t * 16 + rowsel * 8 + e;
                out[(size_t)crow * D_MODEL + ccol] = acc[t][u][e] + bb;
            }
        }
    }
}

// ---------------------------------------------------------------------------
// Kernel 4: per-position attention + residual
// ---------------------------------------------------------------------------
__global__ __launch_bounds__(64)
void attn_kernel(const float* __restrict__ qkv,
                 const float* __restrict__ inp,
                 float* __restrict__ outp) {
    const int pos = blockIdx.x;
    const int i   = threadIdx.x;      // 0..63

    const float* Qg = qkv + (size_t)pos * D_MODEL;
    const float* Kg = qkv + (size_t)ROWS * D_MODEL + (size_t)pos * D_MODEL;
    const float* Vg = qkv + (size_t)2 * ROWS * D_MODEL + (size_t)pos * D_MODEL;

    __shared__ float Kl[64][NH + 1];
    __shared__ float Vl[64][NH + 1];

    float q[NH];
#pragma unroll
    for (int h = 0; h < NH; ++h) {
        q[h]     = Qg[i * NH + h];
        Kl[i][h] = Kg[i * NH + h];
        Vl[i][h] = Vg[i * NH + h];
    }
    __syncthreads();

    const float scale = 0.125f;       // 1/sqrt(DK)

    float mmax = -3.402823466e38f, ssum = 0.f;
    for (int j = 0; j < 64; ++j) {
        float d = 0.f;
#pragma unroll
        for (int h = 0; h < NH; ++h) d += q[h] * Kl[j][h];
        d *= scale;
        const float nm = fmaxf(mmax, d);
        ssum = ssum * __expf(mmax - nm) + __expf(d - nm);
        mmax = nm;
    }
    const float inv = 1.0f / ssum;

    float acc[NH] = {};
    for (int j = 0; j < 64; ++j) {
        float d = 0.f;
#pragma unroll
        for (int h = 0; h < NH; ++h) d += q[h] * Kl[j][h];
        const float wgt = __expf(d * scale - mmax) * inv;
#pragma unroll
        for (int h = 0; h < NH; ++h) acc[h] += wgt * Vl[j][h];
    }

    const float* xi = inp  + (size_t)pos * D_MODEL + i * NH;
    float*       yo = outp + (size_t)pos * D_MODEL + i * NH;
#pragma unroll
    for (int h = 0; h < NH; ++h) yo[h] = xi[h] + acc[h];
}

// ---------------------------------------------------------------------------
// Launch
// ---------------------------------------------------------------------------
extern "C" void kernel_launch(void* const* d_in, const int* in_sizes, int n_in,
                              void* d_out, int out_size, void* d_ws, size_t ws_size,
                              hipStream_t stream) {
    (void)in_sizes; (void)n_in; (void)out_size; (void)ws_size;

    const float* inp   = (const float*)d_in[0];
    const float* Wq    = (const float*)d_in[1];
    const float* bq    = (const float*)d_in[2];
    const float* Wk    = (const float*)d_in[3];
    const float* bk    = (const float*)d_in[4];
    const float* Wv    = (const float*)d_in[5];
    const float* bv    = (const float*)d_in[6];
    const float* gamma = (const float*)d_in[7];
    const float* beta  = (const float*)d_in[8];

    // workspace layout
    const size_t XN_BYTES = (size_t)ROWS * D_MODEL * 2;            // 16 MB bf16
    const size_t WT_BYTES = (size_t)3 * D_MODEL * D_MODEL * 2;     //  6 MB bf16
    unsigned short* xn  = (unsigned short*)d_ws;
    unsigned short* wt  = (unsigned short*)((char*)d_ws + XN_BYTES);
    float*          qkv = (float*)((char*)d_ws + XN_BYTES + WT_BYTES); // 96 MB f32

    ln_bf16_kernel<<<ROWS, 256, 0, stream>>>(inp, gamma, beta, xn);
    wconv_kernel<<<dim3(32, 32, 3), dim3(32, 8), 0, stream>>>(Wq, Wk, Wv, wt);
    qkv_gemm_kernel<<<dim3(D_MODEL / 128, ROWS / 128, 3), 256, 0, stream>>>(
        xn, wt, bq, bk, bv, qkv);
    attn_kernel<<<ROWS, 64, 0, stream>>>(qkv, inp, (float*)d_out);
}
